// MultiHeadAttention_52965536694362
// MI455X (gfx1250) — matrix operand order repaired and checked by reading, not compiled, not run on hardware
//
#include <hip/hip_runtime.h>
#include <hip/hip_bf16.h>

typedef __bf16 bf16;
typedef __attribute__((ext_vector_type(16))) __bf16 bf16x16;
typedef __attribute__((ext_vector_type(8)))  float   f32x8;

constexpr int kB  = 2;
constexpr int kS  = 2048;
constexpr int kD  = 1024;
constexpr int kH  = 16;
constexpr int kHD = 64;
constexpr int kM  = kB * kS;          // 4096 GEMM rows
constexpr float kNeg = -1000000.0f;   // reference mask value

union Frag {
  bf16x16  v;
  uint4    q[2];
  unsigned u[8];
};

static __device__ __forceinline__ f32x8 wmma_bf16(bf16x16 a, bf16x16 b, f32x8 c) {
  return __builtin_amdgcn_wmma_f32_16x16x32_bf16(false, a, false, b, (short)0, c,
                                                 false, false);
}

// A-operand 16x32 bf16 from row-major storage (stride in elements):
// lane l -> row (l&15); K elems 16g..16g+15 contiguous, g = l>>4
static __device__ __forceinline__ bf16x16 load_afrag(const bf16* base, int stride,
                                                     int lane) {
  const bf16* row = base + (size_t)(lane & 15) * stride + ((lane >> 4) << 4);
  Frag f;
  f.q[0] = *(const uint4*)(row);
  f.q[1] = *(const uint4*)(row + 8);
  return f.v;
}

// B-operand 32x16 bf16 loaded from B^T storage (rows = N, stride in elements):
// lane l -> column (l&15); K elems 16g..16g+15 contiguous (same k order as load_afrag)
static __device__ __forceinline__ bf16x16 load_bfrag(const bf16* baseT, int stride,
                                                     int lane) {
  const bf16* row = baseT + (size_t)(lane & 15) * stride + ((lane >> 4) << 4);
  Frag f;
  f.q[0] = *(const uint4*)(row);
  f.q[1] = *(const uint4*)(row + 8);
  return f.v;
}

static __device__ __forceinline__ unsigned pk_bf16(float a, float b) {
  bf16 lo = (bf16)a, hi = (bf16)b;
  unsigned short ul, uh;
  __builtin_memcpy(&ul, &lo, 2);
  __builtin_memcpy(&uh, &hi, 2);
  return (unsigned)ul | ((unsigned)uh << 16);
}

// xor-16 lane exchange (SWAPX16) via ds_swizzle: offset = (0x10<<10)|0x1f
static __device__ __forceinline__ float swz16f(float x) {
  return __int_as_float(__builtin_amdgcn_ds_swizzle(__float_as_int(x), 0x401F));
}
static __device__ __forceinline__ unsigned swz16u(unsigned x) {
  return (unsigned)__builtin_amdgcn_ds_swizzle((int)x, 0x401F);
}

// ---------------------------------------------------------------------------
// Weight transpose + fp32 -> bf16 convert via 64x64 LDS tile (both global
// accesses coalesced):  WT[n*D + k] = W[k*D + n]
// ---------------------------------------------------------------------------
__global__ __launch_bounds__(256) void wconv(const float* __restrict__ W,
                                             bf16* __restrict__ WT) {
  __shared__ bf16 t[64][72];
  const int k0 = blockIdx.x * 64;
  const int n0 = blockIdx.y * 64;
  const int tx = threadIdx.x & 15;   // 16 x float4 across n
  const int ty = threadIdx.x >> 4;   // 16 rows
#pragma unroll
  for (int i = 0; i < 4; ++i) {
    const int kr = ty + i * 16;
    const float4 f = *(const float4*)(W + (size_t)(k0 + kr) * kD + n0 + tx * 4);
    t[tx * 4 + 0][kr] = (bf16)f.x;
    t[tx * 4 + 1][kr] = (bf16)f.y;
    t[tx * 4 + 2][kr] = (bf16)f.z;
    t[tx * 4 + 3][kr] = (bf16)f.w;
  }
  __syncthreads();
#pragma unroll
  for (int i = 0; i < 4; ++i) {
    const int nr = ty + i * 16;
    *(uint2*)(WT + (size_t)(n0 + nr) * kD + k0 + tx * 4) =
        *(const uint2*)(&t[nr][tx * 4]);
  }
}

// ---------------------------------------------------------------------------
// Tiled WMMA GEMM: C[M=4096, N=1024] = A[4096,1024] * B[1024,1024]
//   B supplied transposed (bf16).  A fp32 (converted while staging) or bf16.
//   MODE 0: C -> bf16 [B,H,S,HD]   (Q, K projections, head split)
//   MODE 1: C -> bf16 [B,H,HD,S]   (V projection, transposed per head)
//   MODE 2: C -> fp32 row-major    (final output projection)
// Block: 256 thr = 8 waves, tile 128x128; each wave 32x64 = 2x4 WMMA tiles.
// LDS double-buffered (ping-pong), one barrier per K-step; next tile's global
// loads issue before the current tile's WMMAs.
// ---------------------------------------------------------------------------
template <int MODE, bool A_IS_BF16>
__global__ __launch_bounds__(256) void gemm128(const void* __restrict__ Ap,
                                               const bf16* __restrict__ BT,
                                               void* __restrict__ Cp,
                                               float cscale) {
  constexpr int LDT = 40;  // padded LDS row stride (80 B): conflict-free b128
  __shared__ __align__(16) bf16 As[2][128 * LDT];
  __shared__ __align__(16) bf16 Bs[2][128 * LDT];

  const int tid  = threadIdx.x;
  const int lane = tid & 31;
  const int wave = tid >> 5;
  const int wm   = wave & 3;   // 4 waves down M (32 rows each)
  const int wn   = wave >> 2;  // 2 waves across N (64 cols each)
  const int m0   = blockIdx.y * 128;
  const int n0   = blockIdx.x * 128;

  f32x8 acc[2][4] = {};

  const int lrow = tid >> 1;        // staging: 2 threads per tile row
  const int lkh  = (tid & 1) * 16;  // 16 k-elems each

  uint4 ra0, ra1, rb0, rb1;
  auto gload = [&](int k0) {
    const bf16* bsrc = BT + (size_t)(n0 + lrow) * kD + k0 + lkh;
    rb0 = *(const uint4*)bsrc;
    rb1 = *(const uint4*)(bsrc + 8);
    if (A_IS_BF16) {
      const bf16* asrc = (const bf16*)Ap + (size_t)(m0 + lrow) * kD + k0 + lkh;
      ra0 = *(const uint4*)asrc;
      ra1 = *(const uint4*)(asrc + 8);
    } else {
      const float* asrc = (const float*)Ap + (size_t)(m0 + lrow) * kD + k0 + lkh;
      const float4 f0 = *(const float4*)(asrc + 0);
      const float4 f1 = *(const float4*)(asrc + 4);
      const float4 f2 = *(const float4*)(asrc + 8);
      const float4 f3 = *(const float4*)(asrc + 12);
      ra0.x = pk_bf16(f0.x, f0.y); ra0.y = pk_bf16(f0.z, f0.w);
      ra0.z = pk_bf16(f1.x, f1.y); ra0.w = pk_bf16(f1.z, f1.w);
      ra1.x = pk_bf16(f2.x, f2.y); ra1.y = pk_bf16(f2.z, f2.w);
      ra1.z = pk_bf16(f3.x, f3.y); ra1.w = pk_bf16(f3.z, f3.w);
    }
  };
  auto lstore = [&](int buf) {
    *(uint4*)(As[buf] + lrow * LDT + lkh)     = ra0;
    *(uint4*)(As[buf] + lrow * LDT + lkh + 8) = ra1;
    *(uint4*)(Bs[buf] + lrow * LDT + lkh)     = rb0;
    *(uint4*)(Bs[buf] + lrow * LDT + lkh + 8) = rb1;
  };

  gload(0);
  lstore(0);
  int buf = 0;
  for (int k0 = 0; k0 < kD; k0 += 32) {
    __syncthreads();
    const bool more = (k0 + 32) < kD;
    if (more) gload(k0 + 32);

    bf16x16 af[2], bfr[4];
    af[0] = load_afrag(As[buf] + (wm * 32 + 0)  * LDT, LDT, lane);
    af[1] = load_afrag(As[buf] + (wm * 32 + 16) * LDT, LDT, lane);
#pragma unroll
    for (int j = 0; j < 4; ++j)
      bfr[j] = load_bfrag(Bs[buf] + (wn * 64 + j * 16) * LDT, LDT, lane);
#pragma unroll
    for (int i = 0; i < 2; ++i)
#pragma unroll
      for (int j = 0; j < 4; ++j)
        acc[i][j] = wmma_bf16(af[i], bfr[j], acc[i][j]);

    if (more) lstore(buf ^ 1);
    buf ^= 1;
  }

  const int g  = lane >> 4;
  const int ln = lane & 15;
#pragma unroll
  for (int i = 0; i < 2; ++i)
#pragma unroll
    for (int j = 0; j < 4; ++j)
#pragma unroll
      for (int r = 0; r < 8; ++r) {
        const int gm = m0 + wm * 32 + i * 16 + g * 8 + r;  // C: M = r + 8g
        const int gn = n0 + wn * 64 + j * 16 + ln;         // C: N = lane&15
        const float vv = acc[i][j][r] * cscale;
        if (MODE == 2) {
          ((float*)Cp)[(size_t)gm * kD + gn] = vv;
        } else {
          const int bb = gm >> 11, s = gm & (kS - 1);
          const int hh = gn >> 6, hd = gn & (kHD - 1);
          size_t idx;
          if (MODE == 0)
            idx = (((size_t)bb * kH + hh) * kS + s) * kHD + hd;
          else
            idx = (((size_t)bb * kH + hh) * kHD + hd) * kS + s;
          ((bf16*)Cp)[idx] = (bf16)vv;
        }
      }
}

// ---------------------------------------------------------------------------
// Flash attention key-block body.  Scores computed TRANSPOSED
// (S^T = K_blk * Q^T) so that exp(S^T) in the WMMA C layout feeds the
// B-operand of ctx^T += V^T * P^T with only a lane^16 packed swap.
// MASK=true only for the single block straddling valid_len (1/sqrt(HD) is
// pre-folded into Q, exact in bf16).
// ---------------------------------------------------------------------------
template <bool MASK>
static __device__ __forceinline__ void fa_block(
    const bf16* __restrict__ Kb, const bf16* __restrict__ Vb,
    const bf16x16 (&qf)[2][2], int kb, int vlen, int g, int lane,
    float (&mrun)[2], float (&lrun)[2], f32x8 (&o)[4][2]) {
  bf16x16 kf0[2], kf1[2];
#pragma unroll
  for (int fm = 0; fm < 2; ++fm) {
    kf0[fm] = load_afrag(Kb + (size_t)(kb + fm * 16) * kHD + 0,  kHD, lane);
    kf1[fm] = load_afrag(Kb + (size_t)(kb + fm * 16) * kHD + 32, kHD, lane);
  }
  f32x8 st[2][2] = {};  // S^T[k=32][q=32]
#pragma unroll
  for (int fm = 0; fm < 2; ++fm)
#pragma unroll
    for (int fn = 0; fn < 2; ++fn) {
      st[fm][fn] = wmma_bf16(kf0[fm], qf[0][fn], st[fm][fn]);
      st[fm][fn] = wmma_bf16(kf1[fm], qf[1][fn], st[fm][fn]);
    }
  if (MASK) {
    // valid iff k = kb + 16fm + 8g + r < vlen  <=>  16fm + r < base
    const int base = vlen - kb - g * 8;
#pragma unroll
    for (int fm = 0; fm < 2; ++fm)
#pragma unroll
      for (int fn = 0; fn < 2; ++fn)
#pragma unroll
        for (int r = 0; r < 8; ++r)
          st[fm][fn][r] = (fm * 16 + r < base) ? st[fm][fn][r] : kNeg;
  }
  // online softmax along k for each q column (fn*16 + lane&15)
  float cf[2];
#pragma unroll
  for (int fn = 0; fn < 2; ++fn) {
    float mx = st[0][fn][0];
#pragma unroll
    for (int fm = 0; fm < 2; ++fm)
#pragma unroll
      for (int r = 0; r < 8; ++r) mx = fmaxf(mx, st[fm][fn][r]);
    mx = fmaxf(mx, swz16f(mx));  // partner half-wave holds the other k's
    const float nm = fmaxf(mrun[fn], mx);
    cf[fn]   = __expf(mrun[fn] - nm);
    mrun[fn] = nm;
    float sum = 0.0f;
#pragma unroll
    for (int fm = 0; fm < 2; ++fm)
#pragma unroll
      for (int r = 0; r < 8; ++r) {
        const float p = __expf(st[fm][fn][r] - nm);
        st[fm][fn][r] = p;
        sum += p;
      }
    sum += swz16f(sum);
    lrun[fn] = lrun[fn] * cf[fn] + sum;
  }
#pragma unroll
  for (int fh = 0; fh < 4; ++fh)
#pragma unroll
    for (int fn = 0; fn < 2; ++fn)
#pragma unroll
      for (int r = 0; r < 8; ++r) o[fh][fn][r] *= cf[fn];

  // Build P^T B-frags in registers. B needs k = {16g..16g+15}; locally we
  // hold k = {8g+r} U {16+8g+r}: swap one packed half with lane^16.
  bf16x16 pf[2];
#pragma unroll
  for (int fn = 0; fn < 2; ++fn) {
    unsigned lo[4], hi[4], plo[4], phi[4];
#pragma unroll
    for (int j = 0; j < 4; ++j) {
      lo[j] = pk_bf16(st[0][fn][2 * j], st[0][fn][2 * j + 1]);
      hi[j] = pk_bf16(st[1][fn][2 * j], st[1][fn][2 * j + 1]);
    }
#pragma unroll
    for (int j = 0; j < 4; ++j) {
      plo[j] = swz16u(lo[j]);
      phi[j] = swz16u(hi[j]);
    }
    Frag f;
#pragma unroll
    for (int j = 0; j < 4; ++j) {
      f.u[j]     = g ? phi[j] : lo[j];   // k 16..23 | 0..7
      f.u[4 + j] = g ? hi[j]  : plo[j];  // k 24..31 | 8..15
    }
    pf[fn] = f.v;
  }
  // ctx^T += V^T_blk(64 x 32) * P^T(32 x 32)
#pragma unroll
  for (int fh = 0; fh < 4; ++fh) {
    const bf16x16 vf = load_afrag(Vb + (size_t)(fh * 16) * kS + kb, kS, lane);
#pragma unroll
    for (int fn = 0; fn < 2; ++fn) o[fh][fn] = wmma_bf16(vf, pf[fn], o[fh][fn]);
  }
}

// ---------------------------------------------------------------------------
// Flash attention per (b,h): grid (S/256, B*H), 8 waves x 32 queries each.
// ---------------------------------------------------------------------------
__global__ __launch_bounds__(256) void flash_attn(const bf16* __restrict__ Qh,
                                                  const bf16* __restrict__ Kh,
                                                  const bf16* __restrict__ Vt,
                                                  const int* __restrict__ vlens,
                                                  bf16* __restrict__ ctx) {
  const int lane = threadIdx.x & 31;
  const int wave = threadIdx.x >> 5;
  const int g    = lane >> 4;
  const int ln   = lane & 15;
  const int bh   = blockIdx.y;
  const int b    = bh >> 4;        // / H
  const int h    = bh & (kH - 1);
  const int qbase = blockIdx.x * 256 + wave * 32;
  const bf16* Qb = Qh + (size_t)bh * kS * kHD;
  const bf16* Kb = Kh + (size_t)bh * kS * kHD;
  const bf16* Vb = Vt + (size_t)bh * kHD * kS;
  const int vlen = vlens[b];

  // Q as B-operand: qf[ks][fn], K-dim = hd in [32ks,32ks+32), cols q = fn*16+ln
  bf16x16 qf[2][2];
#pragma unroll
  for (int ks = 0; ks < 2; ++ks)
#pragma unroll
    for (int fn = 0; fn < 2; ++fn)
      qf[ks][fn] =
          load_bfrag(Qb + (size_t)(qbase + fn * 16) * kHD + ks * 32, kHD, lane);

  float mrun[2] = {-__builtin_inff(), -__builtin_inff()};
  float lrun[2] = {0.0f, 0.0f};
  f32x8 o[4][2] = {};  // ctx^T accumulators: [hd frag][q frag]

  const int vclamp = vlen < kS ? vlen : kS;
  const int vfull  = vclamp & ~31;  // fully-valid key blocks: no masking
  for (int kb = 0; kb < vfull; kb += 32)
    fa_block<false>(Kb, Vb, qf, kb, vclamp, g, lane, mrun, lrun, o);
  if (vfull < vclamp)
    fa_block<true>(Kb, Vb, qf, vfull, vclamp, g, lane, mrun, lrun, o);
  // key blocks beyond vclamp contribute exactly 0 (exp(-1e6 - m) underflows)

  // normalize, pack bf16, write ctx as [B,S,D] (heads re-interleaved)
#pragma unroll
  for (int fn = 0; fn < 2; ++fn) {
    const float rl = 1.0f / lrun[fn];
    const int qg = qbase + fn * 16 + ln;
#pragma unroll
    for (int fh = 0; fh < 4; ++fh) {
      uint4 pk;
      pk.x = pk_bf16(o[fh][fn][0] * rl, o[fh][fn][1] * rl);
      pk.y = pk_bf16(o[fh][fn][2] * rl, o[fh][fn][3] * rl);
      pk.z = pk_bf16(o[fh][fn][4] * rl, o[fh][fn][5] * rl);
      pk.w = pk_bf16(o[fh][fn][6] * rl, o[fh][fn][7] * rl);
      bf16* dst = ctx + ((size_t)b * kS + qg) * kD + h * kHD + fh * 16 + g * 8;
      *(uint4*)dst = pk;
    }
  }
}

// ---------------------------------------------------------------------------
extern "C" void kernel_launch(void* const* d_in, const int* in_sizes, int n_in,
                              void* d_out, int out_size, void* d_ws, size_t ws_size,
                              hipStream_t stream) {
  (void)in_sizes; (void)n_in; (void)out_size; (void)ws_size;
  const float* q  = (const float*)d_in[0];
  const float* k  = (const float*)d_in[1];
  const float* v  = (const float*)d_in[2];
  const int*   vl = (const int*)d_in[3];
  const float* Wq = (const float*)d_in[4];
  const float* Wk = (const float*)d_in[5];
  const float* Wv = (const float*)d_in[6];
  const float* Wo = (const float*)d_in[7];

  bf16* ws = (bf16*)d_ws;
  const size_t WW = (size_t)kD * kD;       // 1M elems per weight
  const size_t XX = (size_t)kB * kS * kD;  // 4M elems per activation
  bf16* WqT = ws;
  bf16* WkT = WqT + WW;
  bf16* WvT = WkT + WW;
  bf16* WoT = WvT + WW;
  bf16* Qh  = WoT + WW;   // [B,H,S,HD]  (pre-scaled by 1/sqrt(HD))
  bf16* Kh  = Qh + XX;    // [B,H,S,HD]
  bf16* Vtr = Kh + XX;    // [B,H,HD,S]
  bf16* Cx  = Vtr + XX;   // [B,S,D]

  const dim3 tb(256);
  const dim3 wg(kD / 64, kD / 64);
  wconv<<<wg, tb, 0, stream>>>(Wq, WqT);
  wconv<<<wg, tb, 0, stream>>>(Wk, WkT);
  wconv<<<wg, tb, 0, stream>>>(Wv, WvT);
  wconv<<<wg, tb, 0, stream>>>(Wo, WoT);

  const dim3 gg(kD / 128, kM / 128);  // (8, 32)
  gemm128<0, false><<<gg, tb, 0, stream>>>(q, WqT, Qh, 0.125f);  // fold 1/sqrt(HD)
  gemm128<0, false><<<gg, tb, 0, stream>>>(k, WkT, Kh, 1.0f);
  gemm128<1, false><<<gg, tb, 0, stream>>>(v, WvT, Vtr, 1.0f);

  flash_attn<<<dim3(kS / 256, kB * kH), tb, 0, stream>>>(Qh, Kh, Vtr, vl, Cx);

  gemm128<2, true><<<gg, tb, 0, stream>>>(Cx, WoT, d_out, 1.0f);
}
